// LSTM_DE_44968307589660
// MI455X (gfx1250) — matrix operand order repaired and checked by reading, not compiled, not run on hardware
//
#include <hip/hip_runtime.h>

#define HID 1024
#define BAT 1024
#define NSTEPS 32
#define G4 (4 * HID)

typedef __attribute__((ext_vector_type(16))) __bf16 bf16x16;
typedef __attribute__((ext_vector_type(8)))  __bf16 bf16x8;
typedef __attribute__((ext_vector_type(8)))  float  f32x8;

// async-to-LDS builtin operand types: pointer to <4 x i32> in AS1 / AS3
typedef int v4i __attribute__((vector_size(16)));
typedef __attribute__((address_space(1))) v4i g_v4i;   // global
typedef __attribute__((address_space(3))) v4i l_v4i;   // LDS

__device__ __forceinline__ __bf16 cvt_bf16(float f) {
    union { float f; unsigned u; } v; v.f = f;
    unsigned r = v.u + 0x7FFFu + ((v.u >> 16) & 1u);  // round-to-nearest-even
    union { unsigned short s; __bf16 b; } o; o.s = (unsigned short)(r >> 16);
    return o.b;
}

__device__ __forceinline__ float sigmoid_f(float x) { return 1.0f / (1.0f + __expf(-x)); }
__device__ __forceinline__ float tanh_f(float x)    { return 2.0f / (1.0f + __expf(-2.0f * x)) - 1.0f; }

// ---------------------------------------------------------------------------
// Prep: W_hh fp32 -> bf16, h0 -> bf16 buffer, c0 -> fp32 buffer, bias = b_ih+b_hh,
//       x = in_data.  One thread per W_hh element (4M threads).
// ---------------------------------------------------------------------------
__global__ void __launch_bounds__(256) lstm_prep(
    const float* __restrict__ Whh, const float* __restrict__ h0,
    const float* __restrict__ c0,  const float* __restrict__ bih,
    const float* __restrict__ bhh, const float* __restrict__ in_data,
    __bf16* __restrict__ whh_bf, __bf16* __restrict__ h_bf,
    float* __restrict__ c, float* __restrict__ bias, float* __restrict__ x) {
    int i = blockIdx.x * blockDim.x + threadIdx.x;
    if (i < G4 * HID) whh_bf[i] = cvt_bf16(Whh[i]);
    if (i < BAT * HID) { h_bf[i] = cvt_bf16(h0[i]); c[i] = c0[i]; }
    if (i < G4) bias[i] = bih[i] + bhh[i];
    if (i < BAT) x[i] = in_data[i];
}

// ---------------------------------------------------------------------------
// One LSTM step.  Grid: 128 blocks (8 batch-tiles x 16 j-tiles), 256 threads.
// Block computes gates for 128 batch rows x 64 hidden cols x 4 gates using
// v_wmma_f32_16x16x32_bf16.  W_hh panels stream into double-buffered LDS via
// GLOBAL_LOAD_ASYNC_TO_LDS_B128 (ASYNCcnt); B fragments rotate through THREE
// register slots so two tiles' ds_loads are in flight under each WMMA.
// ---------------------------------------------------------------------------
__global__ void __launch_bounds__(256) lstm_step(
    const __bf16* __restrict__ whh_bf, const __bf16* __restrict__ h_in,
    const float* __restrict__ c_in,    const float* __restrict__ x,
    const float* __restrict__ Wih,     const float* __restrict__ bias,
    __bf16* __restrict__ h_out,        float* __restrict__ c_out) {
    __shared__ __bf16 Bs[2][256 * 32];   // 2 x (256 gate-rows x 32 k) = 32 KB

    const int lane  = threadIdx.x & 31;
    const int wave  = threadIdx.x >> 5;
    const int bTile = blockIdx.x >> 4;   // 0..7
    const int jTile = blockIdx.x & 15;   // 0..15
    const int bBase = bTile * 128 + wave * 16;   // this wave's 16 batch rows
    const int jBase = jTile * 64;

    const int nl = lane & 15;
    const int hi = lane >> 4;            // lane half select
    const int mA     = bBase + nl;       // A-matrix row for this lane
    const int kA_off = hi * 16;          // A lane layout: K 0..15 / 16..31
    const int kB_off = hi * 16;          // B lane layout: K 0..15 / 16..31

    f32x8 acc[4][4];
#pragma unroll
    for (int s = 0; s < 4; ++s)
#pragma unroll
        for (int jt = 0; jt < 4; ++jt) acc[s][jt] = {};

    // Async staging coordinates: thread t owns gate-row (t/64)*H + jBase + t%64,
    // copying 64 B (32 bf16) per panel as 4 x b128 async LDS copies.
    const int sFill = threadIdx.x >> 6;
    const int jFill = threadIdx.x & 63;
    const __bf16* fillBase = whh_bf + (size_t)(sFill * HID + jBase + jFill) * HID;
    __bf16* ldsRow0 = &Bs[0][threadIdx.x * 32];
    __bf16* ldsRow1 = &Bs[1][threadIdx.x * 32];

    // prologue: stream panel 0 into buffer 0
#pragma unroll
    for (int q = 0; q < 4; ++q)
        __builtin_amdgcn_global_load_async_to_lds_b128(
            (g_v4i*)(fillBase + q * 8), (l_v4i*)(ldsRow0 + q * 8), 0, 0);

    for (int kc = 0; kc < HID; kc += 32) {
        const int buf = (kc >> 5) & 1;
        __bf16* ldsNext = buf ? ldsRow0 : ldsRow1;

        // my async copies into Bs[buf] are done; barrier makes everyone's
        // copies visible and guarantees Bs[buf^1] is no longer being read.
        asm volatile("s_wait_asynccnt 0" ::: "memory");
        __syncthreads();

        // stream the next panel into the other buffer (no wait here)
        if (kc + 32 < HID) {
            const __bf16* src = fillBase + kc + 32;
#pragma unroll
            for (int q = 0; q < 4; ++q)
                __builtin_amdgcn_global_load_async_to_lds_b128(
                    (g_v4i*)(src + q * 8), (l_v4i*)(ldsNext + q * 8), 0, 0);
        }

        // ---- A fragment: this wave's 16 h-rows, k-chunk ----
        union { bf16x16 v; bf16x8 h8[2]; } a;
        const __bf16* hp = h_in + (size_t)mA * HID + kc + kA_off;
        a.h8[0] = *(const bf16x8*)hp;
        a.h8[1] = *(const bf16x8*)(hp + 8);
        if (kc + 32 < HID)
            __builtin_prefetch((const void*)(hp + 32), 0, 0);

        // ---- 16 WMMAs: 4 gates x 4 j-tiles, consuming Bs[buf].
        //      B fragments rotate through 3 slots: while WMMA idx runs, the
        //      ds_loads for tiles idx+1 and idx+2 are both in flight. ----
        const __bf16* panel = &Bs[buf][0];
        bf16x16 frag[3];
        frag[0] = *(const bf16x16*)(panel + (0 * 16 + nl) * 32 + kB_off);  // idx 0
        frag[1] = *(const bf16x16*)(panel + (1 * 16 + nl) * 32 + kB_off);  // idx 1
#pragma unroll
        for (int idx = 0; idx < 16; ++idx) {
            if (idx + 2 < 16) {
                const int nidx = idx + 2;
                const int locRow = (nidx >> 2) * 64 + (nidx & 3) * 16 + nl;
                frag[nidx % 3] = *(const bf16x16*)(panel + locRow * 32 + kB_off);
            }
            acc[idx >> 2][idx & 3] = __builtin_amdgcn_wmma_f32_16x16x32_bf16(
                false, a.v, false, frag[idx % 3], (short)0,
                acc[idx >> 2][idx & 3], false, false);
        }
    }

    // ---- elementwise LSTM cell; D layout: (M = r + hi*8, N = nl) per VGPR r ----
#pragma unroll
    for (int r = 0; r < 8; ++r) {
        const int m = bBase + r + hi * 8;
        const float xv = x[m];
#pragma unroll
        for (int jt = 0; jt < 4; ++jt) {
            const int j = jBase + jt * 16 + nl;
            float gi = acc[0][jt][r] + xv * Wih[j]           + bias[j];
            float gf = acc[1][jt][r] + xv * Wih[HID + j]     + bias[HID + j];
            float gg = acc[2][jt][r] + xv * Wih[2 * HID + j] + bias[2 * HID + j];
            float go = acc[3][jt][r] + xv * Wih[3 * HID + j] + bias[3 * HID + j];
            float iv = sigmoid_f(gi);
            float fv = sigmoid_f(gf);
            float gv = tanh_f(gg);
            float ov = sigmoid_f(go);
            float cn = fv * c_in[(size_t)m * HID + j] + iv * gv;
            float hn = ov * tanh_f(cn);
            c_out[(size_t)m * HID + j] = cn;
            h_out[(size_t)m * HID + j] = cvt_bf16(hn);
        }
    }
}

// ---------------------------------------------------------------------------
// Output projection per step: y[b] = h_new[b,:] . W_out + b_out
// Grid: 1024 blocks (one per batch row), 256 threads.  Deterministic reduce.
// ---------------------------------------------------------------------------
__global__ void __launch_bounds__(256) lstm_out(
    const __bf16* __restrict__ h_bf, const float* __restrict__ Wout,
    const float* __restrict__ bout,  float* __restrict__ out,
    float* __restrict__ x, int t) {
    __shared__ float red[8];
    const int b = blockIdx.x;
    float s = 0.0f;
#pragma unroll
    for (int k = threadIdx.x; k < HID; k += 256)
        s += (float)h_bf[(size_t)b * HID + k] * Wout[k];
#pragma unroll
    for (int off = 16; off > 0; off >>= 1) s += __shfl_down(s, off, 32);
    if ((threadIdx.x & 31) == 0) red[threadIdx.x >> 5] = s;
    __syncthreads();
    if (threadIdx.x == 0) {
        float y = bout[0];
#pragma unroll
        for (int w = 0; w < 8; ++w) y += red[w];
        out[b * NSTEPS + t] = y;
        x[b] = y;
    }
}

// ---------------------------------------------------------------------------
extern "C" void kernel_launch(void* const* d_in, const int* in_sizes, int n_in,
                              void* d_out, int out_size, void* d_ws, size_t ws_size,
                              hipStream_t stream) {
    const float* in_data = (const float*)d_in[0];
    const float* h0      = (const float*)d_in[2];
    const float* c0      = (const float*)d_in[3];
    const float* Wih     = (const float*)d_in[4];
    const float* Whh     = (const float*)d_in[5];
    const float* bih     = (const float*)d_in[6];
    const float* bhh     = (const float*)d_in[7];
    const float* Wout    = (const float*)d_in[8];
    const float* bout    = (const float*)d_in[9];
    float* out = (float*)d_out;

    char* ws = (char*)d_ws;
    size_t off = 0;
    auto carve = [&](size_t bytes) {
        char* p = ws + off;
        off = (off + bytes + 255) & ~(size_t)255;
        return p;
    };
    __bf16* whh_bf = (__bf16*)carve((size_t)G4 * HID * sizeof(__bf16)); // 8 MB
    __bf16* hA     = (__bf16*)carve((size_t)BAT * HID * sizeof(__bf16)); // 2 MB
    __bf16* hB     = (__bf16*)carve((size_t)BAT * HID * sizeof(__bf16)); // 2 MB
    float*  cA     = (float*) carve((size_t)BAT * HID * sizeof(float));  // 4 MB
    float*  cB     = (float*) carve((size_t)BAT * HID * sizeof(float));  // 4 MB
    float*  bias   = (float*) carve((size_t)G4 * sizeof(float));
    float*  x      = (float*) carve((size_t)BAT * sizeof(float));
    (void)ws_size; (void)in_sizes; (void)n_in; (void)out_size;

    lstm_prep<<<(G4 * HID + 255) / 256, 256, 0, stream>>>(
        Whh, h0, c0, bih, bhh, in_data, whh_bf, hA, cA, bias, x);

    for (int t = 0; t < NSTEPS; ++t) {
        __bf16* hi = (t & 1) ? hB : hA;
        __bf16* ho = (t & 1) ? hA : hB;
        float*  ci = (t & 1) ? cB : cA;
        float*  co = (t & 1) ? cA : cB;
        lstm_step<<<128, 256, 0, stream>>>(whh_bf, hi, ci, x, Wih, bias, ho, co);
        lstm_out<<<BAT, 256, 0, stream>>>(ho, Wout, bout, out, x, t);
    }
}
